// LSTM_54339926229024
// MI455X (gfx1250) — matrix operand order repaired and checked by reading, not compiled, not run on hardware
//
#include <hip/hip_runtime.h>
#include <hip/hip_bf16.h>
#include <math.h>

// ---------------------------------------------------------------------------
// LSTM  (B=128, T=512, IN=512, HID=1024, OUT=256)  for MI455X / gfx1250
//
// Compute-bound (AI ~ 4000 FLOP/byte of HBM): all matmuls go through
// v_wmma_f32_16x16x32_bf16 (fp32 accumulate).  Weights are packed ONCE into
// per-lane 32B WMMA B-fragments (bf16).  The step GEMM register-blocks
// 2 M-tiles per wave so every B fragment read from L2 feeds 2 WMMAs
// (0.5 KB global + 1 KB LDS per WMMA), with 4 independent accumulator
// chains to hide WMMA latency.
// ---------------------------------------------------------------------------

#define BB   128
#define TT   512
#define INF  512
#define HID  1024
#define OUTF 256
#define KTOT 1536              // IN + HID
#define NG   4096              // 4 gates * HID
#define KT_G 48                // KTOT / 32 k-tiles per gate GEMM
#define KT_F 32                // HID/32 k-tiles for FC

typedef __attribute__((ext_vector_type(16))) __bf16 v16bf;
typedef __attribute__((ext_vector_type(8)))  float  v8f;

union Frag32B { uint4 q[2]; v16bf v; };     // one lane's 32-byte WMMA operand
union Pack8   { __bf16 h[8];  uint4 q; };   // 8 bf16 <-> uint4
union Pack16  { __bf16 h[16]; uint4 q[2]; };

// ---------------------------------------------------------------------------
// One-time weight packing:  W[g] is fp32 [1536,1024] row-major (k*1024+n).
// B-fragment for tile (ntile, kt), lane l:
//   N = ntile*16 + (l&15),  K = kt*32 + (l>>4)*8 + j + (j>=8 ? 8 : 0)   (j = 0..15)
//   stored at ((ntile*48 + kt)*32 + l)*16 bf16 elements.
// ---------------------------------------------------------------------------
__global__ __launch_bounds__(256)
void pack_gate_weights(const float* __restrict__ Wf, const float* __restrict__ Wi,
                       const float* __restrict__ Wc, const float* __restrict__ Wo,
                       __bf16* __restrict__ wq) {
    int fl = blockIdx.x * 256 + threadIdx.x;          // 0 .. 4*64*48*32-1
    int lane  = fl & 31;
    int kt    = (fl >> 5) % KT_G;
    int ntile = fl / (KT_G * 32);                     // 0..255 (gate-major)
    int g     = ntile >> 6;
    const float* W = (g == 0) ? Wf : (g == 1) ? Wi : (g == 2) ? Wc : Wo;
    int ncol  = (ntile & 63) * 16 + (lane & 15);
    int kbase = kt * 32 + (lane >> 4) * 8;
    Pack16 p;
#pragma unroll
    for (int j = 0; j < 16; ++j)
        p.h[j] = (__bf16)W[(size_t)(kbase + j + (j >= 8 ? 8 : 0)) * HID + ncol];
    uint4* dst = (uint4*)(wq + ((size_t)(ntile * KT_G + kt) * 32 + lane) * 16);
    dst[0] = p.q[0];
    dst[1] = p.q[1];
}

__global__ __launch_bounds__(256)
void pack_fc_weights(const float* __restrict__ Wfc, __bf16* __restrict__ wq) {
    int fl = blockIdx.x * 256 + threadIdx.x;          // 0 .. 16*32*32-1
    int lane  = fl & 31;
    int kt    = (fl >> 5) % KT_F;
    int ntile = fl / (KT_F * 32);                     // 0..15
    int ncol  = ntile * 16 + (lane & 15);
    int kbase = kt * 32 + (lane >> 4) * 8;
    Pack16 p;
#pragma unroll
    for (int j = 0; j < 16; ++j)
        p.h[j] = (__bf16)Wfc[(size_t)(kbase + j + (j >= 8 ? 8 : 0)) * OUTF + ncol];
    uint4* dst = (uint4*)(wq + ((size_t)(ntile * KT_F + kt) * 32 + lane) * 16);
    dst[0] = p.q[0];
    dst[1] = p.q[1];
}

__global__ __launch_bounds__(256)
void init_state(float* __restrict__ c, __bf16* __restrict__ h) {
    int i = blockIdx.x * 256 + threadIdx.x;           // 128*1024 threads
    c[i] = 0.0f;
    h[i] = (__bf16)0.0f;
}

// ---------------------------------------------------------------------------
// Per-step gate GEMM:  gates[128,4096] = [x_t | h_prev] @ Wall + bias
// grid (4, 32) x 256 threads.  blockIdx.x = 32-row M group (2 m-tiles),
// wave w of blockIdx.y owns n-tile = blockIdx.y*8 + w for BOTH m-tiles:
// each B fragment loaded from L2 feeds 2 WMMAs.
// A tile (32 x 1536, bf16) staged in LDS in A-fragment layout:
//   lane l: M = l&15, first 8 elems K = kt*32+(l>>4)*8+j, next 8 at K+16.
// ---------------------------------------------------------------------------
__device__ inline uint4 fetch8_bf16(const float* __restrict__ x,
                                    const __bf16* __restrict__ h,
                                    int row, int t, int K0) {
    if (K0 < INF) {
        const float4* p = (const float4*)(x + ((size_t)row * TT + t) * INF + K0);
        float4 a = p[0], b = p[1];
        Pack8 u;
        u.h[0] = (__bf16)a.x; u.h[1] = (__bf16)a.y;
        u.h[2] = (__bf16)a.z; u.h[3] = (__bf16)a.w;
        u.h[4] = (__bf16)b.x; u.h[5] = (__bf16)b.y;
        u.h[6] = (__bf16)b.z; u.h[7] = (__bf16)b.w;
        return u.q;
    }
    return *(const uint4*)(h + (size_t)row * HID + (K0 - INF));
}

__global__ __launch_bounds__(256)
void gemm_gates(const float* __restrict__ x, int t,
                const __bf16* __restrict__ h_prev,
                const __bf16* __restrict__ wq,
                const float* __restrict__ bf_, const float* __restrict__ bi_,
                const float* __restrict__ bc_, const float* __restrict__ bo_,
                float* __restrict__ gates) {
    // fragment index: ((kt*2 + mi)*32 + lane) -> 2 x uint4
    __shared__ uint4 ldsA[KT_G * 2 * 32 * 2];         // 96 KB
    const int tid  = threadIdx.x;
    const int wave = tid >> 5;
    const int lane = tid & 31;
    const int mg   = blockIdx.x;                      // 0..3 (32-row group)
    const int ntile = blockIdx.y * 8 + wave;          // 0..255

    // ---- stage A tiles (2 m-tiles) into LDS ----
#pragma unroll
    for (int e = tid; e < KT_G * 2 * 32; e += 256) {
        int kt = e >> 6, mi = (e >> 5) & 1, l = e & 31;
        int row = mg * 32 + mi * 16 + (l & 15);
        int K0  = kt * 32 + (l >> 4) * 8;
        ldsA[e * 2 + 0] = fetch8_bf16(x, h_prev, row, t, K0);
        ldsA[e * 2 + 1] = fetch8_bf16(x, h_prev, row, t, K0 + 16);
    }
    __syncthreads();

    // ---- 4 accumulator chains: [m-tile 0/1] x [k-parity], bias folded in ----
    const int g = ntile >> 6;
    const float* bias_p = (g == 0) ? bf_ : (g == 1) ? bi_ : (g == 2) ? bc_ : bo_;
    float bias = bias_p[(ntile & 63) * 16 + (lane & 15)];
    v8f acc00, acc01, acc10, acc11;
#pragma unroll
    for (int r = 0; r < 8; ++r) {
        acc00[r] = bias; acc10[r] = bias;
        acc01[r] = 0.0f; acc11[r] = 0.0f;
    }

    const uint4* bbase = (const uint4*)(wq + (size_t)ntile * KT_G * 512 + lane * 16);
#pragma unroll 2
    for (int kt = 0; kt < KT_G; kt += 2) {
        Frag32B A00, A10, A01, A11, B0, B1;
        const uint4* a = &ldsA[(kt * 2 * 32 + lane) * 2];
        A00.q[0] = a[0];        A00.q[1] = a[1];          // kt,   m-tile 0
        A10.q[0] = a[64];       A10.q[1] = a[65];         // kt,   m-tile 1
        A01.q[0] = a[128];      A01.q[1] = a[129];        // kt+1, m-tile 0
        A11.q[0] = a[192];      A11.q[1] = a[193];        // kt+1, m-tile 1
        const uint4* b0 = bbase + kt * 64;                // 1024 B per fragment
        B0.q[0] = b0[0];   B0.q[1] = b0[1];
        B1.q[0] = b0[64];  B1.q[1] = b0[65];
        acc00 = __builtin_amdgcn_wmma_f32_16x16x32_bf16(false, A00.v, false, B0.v,
                                                        (short)0, acc00, false, false);
        acc10 = __builtin_amdgcn_wmma_f32_16x16x32_bf16(false, A10.v, false, B0.v,
                                                        (short)0, acc10, false, false);
        acc01 = __builtin_amdgcn_wmma_f32_16x16x32_bf16(false, A01.v, false, B1.v,
                                                        (short)0, acc01, false, false);
        acc11 = __builtin_amdgcn_wmma_f32_16x16x32_bf16(false, A11.v, false, B1.v,
                                                        (short)0, acc11, false, false);
    }
    acc00 += acc01;
    acc10 += acc11;

    // ---- store: C layout lane 0-15 M=r, lane 16-31 M=8+r; N = lane&15 ----
    int n = ntile * 16 + (lane & 15);
    int mb0 = mg * 32 + (lane >> 4) * 8;
    int mb1 = mb0 + 16;
#pragma unroll
    for (int r = 0; r < 8; ++r)
        gates[(size_t)(mb0 + r) * NG + n] = acc00[r];
#pragma unroll
    for (int r = 0; r < 8; ++r)
        gates[(size_t)(mb1 + r) * NG + n] = acc10[r];
}

// ---------------------------------------------------------------------------
// Per-step elementwise cell update (gate order: f, i, c~, o)
// ---------------------------------------------------------------------------
__device__ inline float sigf(float v) { return 1.0f / (1.0f + __expf(-v)); }

__global__ __launch_bounds__(256)
void lstm_cell(const float* __restrict__ gates, float* __restrict__ c,
               __bf16* __restrict__ h) {
    int idx = blockIdx.x * 256 + threadIdx.x;         // 128*1024
    int m = idx >> 10, n = idx & 1023;
    const float* gm = gates + (size_t)m * NG;
    float f  = sigf(gm[n]);
    float i  = sigf(gm[HID + n]);
    float ct = tanhf(gm[2 * HID + n]);
    float o  = sigf(gm[3 * HID + n]);
    float cn = f * c[idx] + i * ct;
    c[idx] = cn;
    h[idx] = (__bf16)(o * tanhf(cn));
}

// ---------------------------------------------------------------------------
// Final FC: out = LeakyReLU(h @ W_fc + b_fc).  grid (8,2) x 256.
// A fragments loaded straight from h (bf16, contiguous 16B groups).
// ---------------------------------------------------------------------------
__global__ __launch_bounds__(256)
void fc_out(const __bf16* __restrict__ h, const __bf16* __restrict__ wq,
            const float* __restrict__ bfc, float* __restrict__ out) {
    const int tid  = threadIdx.x;
    const int wave = tid >> 5;
    const int lane = tid & 31;
    const int mt   = blockIdx.x;                      // 0..7
    const int nt   = blockIdx.y * 8 + wave;           // 0..15

    float bias = bfc[nt * 16 + (lane & 15)];
    v8f acc0, acc1;
#pragma unroll
    for (int r = 0; r < 8; ++r) { acc0[r] = bias; acc1[r] = 0.0f; }

    const int M = lane & 15, kh = lane >> 4;
    const __bf16* arow = h + (size_t)(mt * 16 + M) * HID;
    const uint4* bbase = (const uint4*)(wq + (size_t)nt * KT_F * 512 + lane * 16);
#pragma unroll 4
    for (int kt = 0; kt < KT_F; kt += 2) {
        Frag32B A0, A1, B0, B1;
        int K0 = kt * 32 + kh * 8;
        A0.q[0] = *(const uint4*)(arow + K0);
        A0.q[1] = *(const uint4*)(arow + K0 + 16);
        A1.q[0] = *(const uint4*)(arow + K0 + 32);
        A1.q[1] = *(const uint4*)(arow + K0 + 48);
        const uint4* b0 = bbase + kt * 64;
        const uint4* b1 = bbase + (kt + 1) * 64;
        B0.q[0] = b0[0]; B0.q[1] = b0[1];
        B1.q[0] = b1[0]; B1.q[1] = b1[1];
        acc0 = __builtin_amdgcn_wmma_f32_16x16x32_bf16(false, A0.v, false, B0.v,
                                                       (short)0, acc0, false, false);
        acc1 = __builtin_amdgcn_wmma_f32_16x16x32_bf16(false, A1.v, false, B1.v,
                                                       (short)0, acc1, false, false);
    }
    acc0 += acc1;

    int n = nt * 16 + (lane & 15);
    int mbase = mt * 16 + (lane >> 4) * 8;
#pragma unroll
    for (int r = 0; r < 8; ++r) {
        float v = acc0[r];
        out[(size_t)(mbase + r) * OUTF + n] = (v >= 0.0f) ? v : 0.01f * v;
    }
}

// ---------------------------------------------------------------------------
// Host driver: pack weights, init state, 512 x (gemm + cell), final FC.
// All launches on `stream` (graph-capture safe, deterministic).
// ---------------------------------------------------------------------------
extern "C" void kernel_launch(void* const* d_in, const int* in_sizes, int n_in,
                              void* d_out, int out_size, void* d_ws, size_t ws_size,
                              hipStream_t stream) {
    const float* x   = (const float*)d_in[0];
    const float* Wf  = (const float*)d_in[1];
    const float* bf_ = (const float*)d_in[2];
    const float* Wi  = (const float*)d_in[3];
    const float* bi_ = (const float*)d_in[4];
    const float* Wc  = (const float*)d_in[5];
    const float* bc_ = (const float*)d_in[6];
    const float* Wo  = (const float*)d_in[7];
    const float* bo_ = (const float*)d_in[8];
    const float* Wfc = (const float*)d_in[9];
    const float* bfc = (const float*)d_in[10];
    float* out = (float*)d_out;

    char* ws = (char*)d_ws;
    __bf16* wq_g  = (__bf16*)(ws);                        // 12,582,912 B
    __bf16* wq_f  = (__bf16*)(ws + 12582912);             //  1,048,576 B
    float*  gates = (float*) (ws + 13631488);             //  2,097,152 B
    float*  cst   = (float*) (ws + 15728640);             //    524,288 B
    __bf16* hst   = (__bf16*)(ws + 16252928);             //    262,144 B

    pack_gate_weights<<<1536, 256, 0, stream>>>(Wf, Wi, Wc, Wo, wq_g);
    pack_fc_weights<<<64, 256, 0, stream>>>(Wfc, wq_f);
    init_state<<<512, 256, 0, stream>>>(cst, hst);

    for (int t = 0; t < TT; ++t) {
        gemm_gates<<<dim3(4, 32), 256, 0, stream>>>(x, t, hst, wq_g,
                                                    bf_, bi_, bc_, bo_, gates);
        lstm_cell<<<512, 256, 0, stream>>>(gates, cst, hst);
    }
    fc_out<<<dim3(8, 2), 256, 0, stream>>>(hst, wq_f, bfc, out);
}
